// DBSE_57861799412215
// MI455X (gfx1250) — compile-verified, matrix operand order checked by reading, not run on hardware
//
#include <hip/hip_runtime.h>
#include <hip/hip_bf16.h>
#include <stdint.h>

// ---------------- model dims ----------------
#define BATCH 128
#define TDIM  128
#define FEAT  1024
#define FCD   512
#define WINW  8
#define ZDD   32
#define FDD   256
#define HIDD  256
#define NWW   16   // TDIM / WINW

#define KC 32      // K chunk per stage (one WMMA K depth)

typedef _Float16 v16h __attribute__((ext_vector_type(16)));
typedef _Float16 v8h  __attribute__((ext_vector_type(8)));
typedef float    v8f  __attribute__((ext_vector_type(8)));
typedef int      v4i  __attribute__((ext_vector_type(4)));

#define AS1 __attribute__((address_space(1)))
#define AS3 __attribute__((address_space(3)))

#if defined(__has_builtin)
#if __has_builtin(__builtin_amdgcn_global_load_async_to_lds_b128)
#define HAVE_ASYNC_LDS 1
#endif
#endif

// ---------------- device helpers ----------------
__device__ __forceinline__ float sigm(float x) { return 1.0f / (1.0f + expf(-x)); }

__device__ __forceinline__ uint32_t hashu(uint32_t x) {
    x ^= x >> 16; x *= 0x7feb352dU;
    x ^= x >> 15; x *= 0x846ca68bU;
    x ^= x >> 16; return x;
}
__device__ __forceinline__ float u01(uint32_t h) {
    return (float)(h >> 8) * (1.0f / 16777216.0f) + (0.5f / 16777216.0f);
}
// deterministic counter-based standard normal (Box-Muller on a hash)
__device__ __forceinline__ float nrm(uint32_t seed, uint32_t idx) {
    uint32_t h1 = hashu(seed * 0x9E3779B9u ^ (idx * 0x85EBCA6Bu + 0x165667B1u));
    uint32_t h2 = hashu(h1 ^ 0xC2B2AE35u);
    float u1 = u01(h1), u2 = u01(h2);
    return sqrtf(-2.0f * logf(u1)) * cosf(6.28318530717958647692f * u2);
}

__device__ __forceinline__ v8f wmma_f16(v16h a, v16h b, v8f c) {
    return __builtin_amdgcn_wmma_f32_16x16x32_f16(false, a, false, b, (short)0, c,
                                                  false, false);
}

// XOR-swizzled LDS offset (halfs) inside a [rows x 32] f16 tile; c = 16B chunk 0..3.
__device__ __forceinline__ int swz(int r, int c) {
    return r * KC + ((c ^ ((r >> 2) & 3)) * 8);
}

// stage one 16-byte chunk global -> LDS (async on gfx1250 toolchains that expose it)
__device__ __forceinline__ void stage16(const _Float16* __restrict__ g, _Float16* l) {
#if defined(HAVE_ASYNC_LDS)
    __builtin_amdgcn_global_load_async_to_lds_b128((v4i*)(void*)g, (AS3 v4i*)l, 0, 0);
#else
    *(v8h*)l = *(const v8h*)g;   // global_load_b128 + ds_store_b128
#endif
}

__device__ __forceinline__ void stage_wait() {
#if defined(HAVE_ASYNC_LDS)
#if __has_builtin(__builtin_amdgcn_s_wait_asynccnt)
    __builtin_amdgcn_s_wait_asynccnt(0);
#else
    asm volatile("s_wait_asynccnt 0" ::: "memory");
#endif
#endif
}

// build one A/B fragment (v16h) from an LDS tile: row r, lane K-half `half`.
// elements [0..7] <- K kb+half*8.., [8..15] <- K kb+16+half*8.. (ISA 7.12.2 layout)
__device__ __forceinline__ v16h frag_lds(const _Float16* t, int r, int half) {
    v8h lo = *(const v8h*)&t[swz(r, half)];
    v8h hi = *(const v8h*)&t[swz(r, half + 2)];
    v16h x;
#pragma unroll
    for (int i = 0; i < 8; ++i) { x[i] = lo[i]; x[i + 8] = hi[i]; }
    return x;
}

__device__ __forceinline__ void epi_store(v8f acc, int rb, int col,
                                          float* __restrict__ C, int ldc,
                                          _Float16* __restrict__ Ch, int ldch,
                                          const float* __restrict__ bias,
                                          int N, int act) {
    if (col >= N) return;
    float bi = bias ? bias[col] : 0.0f;
#pragma unroll
    for (int r = 0; r < 8; ++r) {
        float x = acc[r] + bi;
        if (act == 1) x = tanhf(x);
        if (C)  C [(size_t)(rb + r) * ldc  + col] = x;
        if (Ch) Ch[(size_t)(rb + r) * ldch + col] = (_Float16)x;
    }
}

// ---- WMMA GEMM: D = A[M,K](f16) * Bt[N,K](f16)^T + bias, LDS-staged ----
// Requirements: M % 128 == 0, K % 32 == 0. N arbitrary (OOB rows clamped).
// Block: 256 thr = 8 waves (4 in M x 2 in N); block tile 128M x 64N; wave 32x32.
__global__ __launch_bounds__(256)
void k_wmma_gemm(const _Float16* __restrict__ A, int lda,
                 const _Float16* __restrict__ Bt, int ldb,
                 float* __restrict__ C, int ldc,
                 _Float16* __restrict__ Ch, int ldch,
                 const float* __restrict__ bias,
                 int M, int N, int K, int act) {
    __shared__ __align__(16) _Float16 lA[128 * KC];   // 8 KB
    __shared__ __align__(16) _Float16 lB[64 * KC];    // 4 KB

    const int tid  = threadIdx.x;
    const int lane = tid & 31;
    const int wid  = tid >> 5;
    const int wm0  = (wid & 3) * 32;
    const int wn0  = (wid >> 2) * 32;
    const int bm0  = blockIdx.y * 128;
    const int bn0  = blockIdx.x * 64;
    const int half = lane >> 4;
    const int l15  = lane & 15;

    // per-thread staging coordinates
    const int ar0 = tid >> 2, ac = tid & 3;           // A chunk 0: rows 0..63
    const int ar1 = ar0 + 64;                          // A chunk 1: rows 64..127
    int brg = bn0 + (tid >> 2);                        // B row (clamped)
    if (brg > N - 1) brg = (N > 0) ? N - 1 : 0;
    const _Float16* gA0 = A  + (size_t)(bm0 + ar0) * lda + ac * 8;
    const _Float16* gA1 = A  + (size_t)(bm0 + ar1) * lda + ac * 8;
    const _Float16* gB  = Bt + (size_t)brg * ldb + ac * 8;

    v8f c00 = {}, c01 = {}, c10 = {}, c11 = {};

    for (int kb = 0; kb < K; kb += KC) {
        if (kb + KC < K) {  // gfx1250 global_prefetch_b8 of next K tile
            __builtin_prefetch(gA0 + kb + KC, 0, 0);
            __builtin_prefetch(gB + kb + KC, 0, 0);
        }
        stage16(gA0 + kb, &lA[swz(ar0, ac)]);
        stage16(gA1 + kb, &lA[swz(ar1, ac)]);
        stage16(gB  + kb, &lB[swz(tid >> 2, ac)]);
        stage_wait();
        __syncthreads();

        v16h a0 = frag_lds(lA, wm0 + l15,      half);
        v16h a1 = frag_lds(lA, wm0 + 16 + l15, half);
        v16h b0 = frag_lds(lB, wn0 + l15,      half);
        v16h b1 = frag_lds(lB, wn0 + 16 + l15, half);
        c00 = wmma_f16(a0, b0, c00);
        c01 = wmma_f16(a0, b1, c01);
        c10 = wmma_f16(a1, b0, c10);
        c11 = wmma_f16(a1, b1, c11);
        __syncthreads();
    }

    const int col0 = bn0 + wn0 + l15;
    const int col1 = col0 + 16;
    const int rb0  = bm0 + wm0 + half * 8;
    epi_store(c00, rb0,      col0, C, ldc, Ch, ldch, bias, N, act);
    epi_store(c01, rb0,      col1, C, ldc, Ch, ldch, bias, N, act);
    epi_store(c10, rb0 + 16, col0, C, ldc, Ch, ldch, bias, N, act);
    epi_store(c11, rb0 + 16, col1, C, ldc, Ch, ldch, bias, N, act);
}

// ---------------- elementwise kernels ----------------
__global__ void k_f32_to_f16(const float* __restrict__ in, _Float16* __restrict__ out,
                             int n) {
    int i = blockIdx.x * blockDim.x + threadIdx.x;
    if (i < n) out[i] = (_Float16)in[i];
}

// in [K,N] f32 row-major -> out [N,K] f16
__global__ void k_transpose_f16(const float* __restrict__ in, _Float16* __restrict__ out,
                                int K, int N) {
    int i = blockIdx.x * blockDim.x + threadIdx.x;
    if (i >= K * N) return;
    int k = i / N, n = i % N;
    out[(size_t)n * K + k] = (_Float16)in[i];
}

__global__ void k_rng_normal_f16(_Float16* __restrict__ out, int n, uint32_t seed) {
    int i = blockIdx.x * blockDim.x + threadIdx.x;
    if (i < n) out[i] = (_Float16)nrm(seed, (uint32_t)i);
}

// PyTorch GRU gate math
__global__ void k_gru_gate(const float* __restrict__ gi, const float* __restrict__ gh,
                           const float* __restrict__ hprev, float* __restrict__ h,
                           _Float16* __restrict__ hh, int Bn, int H) {
    int i = blockIdx.x * blockDim.x + threadIdx.x;
    if (i >= Bn * H) return;
    int b = i / H, j = i % H;
    const float* gib = gi + (size_t)b * 3 * H;
    const float* ghb = gh + (size_t)b * 3 * H;
    float r = sigm(gib[j] + ghb[j]);
    float u = sigm(gib[H + j] + ghb[H + j]);
    float n = tanhf(gib[2 * H + j] + r * ghb[2 * H + j]);
    float o = (1.0f - u) * n + u * hprev[i];
    h[i] = o; hh[i] = (_Float16)o;
}

// PyTorch LSTMCell gate math (i,f,g,o chunks)
__global__ void k_lstm_gate(const float* __restrict__ gi, const float* __restrict__ gh,
                            float* __restrict__ c, float* __restrict__ h,
                            _Float16* __restrict__ hh, int Mr, int H) {
    int i = blockIdx.x * blockDim.x + threadIdx.x;
    if (i >= Mr * H) return;
    int b = i / H, j = i % H;
    const float* A = gi + (size_t)b * 4 * H;
    const float* G = gh + (size_t)b * 4 * H;
    float ig = sigm(A[j] + G[j]);
    float fg = sigm(A[H + j] + G[H + j]);
    float gg = tanhf(A[2 * H + j] + G[2 * H + j]);
    float og = sigm(A[3 * H + j] + G[3 * H + j]);
    float cc = fg * c[i] + ig * gg;
    float hv = og * tanhf(cc);
    c[i] = cc; h[i] = hv; hh[i] = (_Float16)hv;
}

// out[r*ld+c] = mean[r*ld+c] + eps*exp(0.5*logvar[r*ld+c]); aliasing out==mean is fine
__global__ void k_sample(const float* __restrict__ mean, const float* __restrict__ logvar,
                         float* __restrict__ out, int rows, int cols, int ld,
                         uint32_t seed) {
    int i = blockIdx.x * blockDim.x + threadIdx.x;
    if (i >= rows * cols) return;
    size_t o = (size_t)(i / cols) * ld + (i % cols);
    out[o] = mean[o] + nrm(seed, (uint32_t)i) * expf(0.5f * logvar[o]);
}

// zdyn: first window rows replaced by ff (repeated), rest = fc_x
__global__ void k_build_zdyn(const _Float16* __restrict__ fcx,
                             const _Float16* __restrict__ ff,
                             _Float16* __restrict__ zdyn, int total) {
    int i = blockIdx.x * blockDim.x + threadIdx.x;
    if (i >= total) return;
    int b = i / (TDIM * FCD);
    int rem = i % (TDIM * FCD);
    int t = rem / FCD, j = rem % FCD;
    zdyn[i] = (t < WINW) ? ff[b * FCD + j] : fcx[i];
}

// zin (f16): teacher-forced shifted z_post
__global__ void k_build_zin(const float* __restrict__ zpost, _Float16* __restrict__ zin,
                            int total) {
    int i = blockIdx.x * blockDim.x + threadIdx.x;
    if (i >= total) return;
    int b = i / (NWW * ZDD);
    int rem = i % (NWW * ZDD);
    int t = rem / ZDD, j = rem % ZDD;
    zin[i] = (t == 0) ? (_Float16)0.0f
                      : (_Float16)zpost[b * NWW * ZDD + (t - 1) * ZDD + j];
}

// zf [B*NW, ZD+FD] = [z_post | f_post broadcast]
__global__ void k_build_zf(const float* __restrict__ zpost,
                           const _Float16* __restrict__ fpost_h,
                           _Float16* __restrict__ zf, int total) {
    int i = blockIdx.x * blockDim.x + threadIdx.x;
    if (i >= total) return;
    int m = i / (ZDD + FDD), j = i % (ZDD + FDD);
    int b = m / NWW;
    zf[i] = (j < ZDD) ? (_Float16)zpost[m * ZDD + j] : fpost_h[b * FDD + (j - ZDD)];
}

// zf_stat [B, ZD+FD] = [z_post[:,0] | f_post]
__global__ void k_build_zf_stat(const float* __restrict__ zpost,
                                const _Float16* __restrict__ fpost_h,
                                _Float16* __restrict__ zf, int total) {
    int i = blockIdx.x * blockDim.x + threadIdx.x;
    if (i >= total) return;
    int b = i / (ZDD + FDD), j = i % (ZDD + FDD);
    zf[i] = (j < ZDD) ? (_Float16)zpost[b * NWW * ZDD + j] : fpost_h[b * FDD + (j - ZDD)];
}

// ---------------- host side ----------------
static inline size_t al256(size_t x) { return (x + 255) & ~(size_t)255; }
static inline int cdiv(int a, int b) { return (a + b - 1) / b; }

extern "C" void kernel_launch(void* const* d_in, const int* in_sizes, int n_in,
                              void* d_out, int out_size, void* d_ws, size_t ws_size,
                              hipStream_t stream) {
    (void)in_sizes; (void)n_in; (void)out_size; (void)ws_size;
    const float* x_seq     = (const float*)d_in[0];
    const float* enc_w     = (const float*)d_in[2];
    const float* enc_b     = (const float*)d_in[3];
    const float* gru_wih   = (const float*)d_in[4];
    const float* gru_whh   = (const float*)d_in[5];
    const float* gru_bih   = (const float*)d_in[6];
    const float* gru_bhh   = (const float*)d_in[7];
    const float* sfe_w     = (const float*)d_in[8];
    const float* sfe_b     = (const float*)d_in[9];
    const float* fmean_w   = (const float*)d_in[10];
    const float* fmean_b   = (const float*)d_in[11];
    const float* flogvar_w = (const float*)d_in[12];
    const float* flogvar_b = (const float*)d_in[13];
    const float* lstm_wih  = (const float*)d_in[14];
    const float* lstm_whh  = (const float*)d_in[15];
    const float* lstm_bih  = (const float*)d_in[16];
    const float* lstm_bhh  = (const float*)d_in[17];
    const float* zmean_w   = (const float*)d_in[18];
    const float* zmean_b   = (const float*)d_in[19];
    const float* zlogvar_w = (const float*)d_in[20];
    const float* zlogvar_b = (const float*)d_in[21];
    const float* p1_wih    = (const float*)d_in[22];
    const float* p1_whh    = (const float*)d_in[23];
    const float* p1_bih    = (const float*)d_in[24];
    const float* p1_bhh    = (const float*)d_in[25];
    const float* p2_wih    = (const float*)d_in[26];
    const float* p2_whh    = (const float*)d_in[27];
    const float* p2_bih    = (const float*)d_in[28];
    const float* p2_bhh    = (const float*)d_in[29];
    const float* pmean_w   = (const float*)d_in[30];
    const float* pmean_b   = (const float*)d_in[31];
    const float* plogvar_w = (const float*)d_in[32];
    const float* plogvar_b = (const float*)d_in[33];
    const float* decm_w    = (const float*)d_in[34];
    const float* decm_b    = (const float*)d_in[35];
    const float* decl_w    = (const float*)d_in[36];
    const float* decl_b    = (const float*)d_in[37];
    float* out = (float*)d_out;

    // ---- output offsets (floats), in reference return order ----
    const size_t o_fmean   = 0;
    const size_t o_flogvar = o_fmean   + (size_t)BATCH * FDD;
    const size_t o_fpost   = o_flogvar + (size_t)BATCH * FDD;
    const size_t o_zmean   = o_fpost   + (size_t)BATCH * FDD;
    const size_t o_zlogvar = o_zmean   + (size_t)BATCH * NWW * ZDD;
    const size_t o_zpost   = o_zlogvar + (size_t)BATCH * NWW * ZDD;
    const size_t o_zpm     = o_zpost   + (size_t)BATCH * NWW * ZDD;
    const size_t o_zplv    = o_zpm     + (size_t)BATCH * NWW * ZDD;
    const size_t o_zp      = o_zplv    + (size_t)BATCH * NWW * ZDD;
    const size_t o_recon   = o_zp      + (size_t)BATCH * NWW * ZDD;
    const size_t o_reconf  = o_recon   + (size_t)BATCH * NWW * FEAT;

    // ---- workspace bump allocator ----
    char* wp = (char*)d_ws;
    auto alloc = [&](size_t bytes) -> void* {
        void* p = (void*)wp; wp += al256(bytes); return p;
    };
    _Float16* xseq_h   = (_Float16*)alloc((size_t)BATCH * TDIM * FEAT * 2);
    _Float16* fcx_h    = (_Float16*)alloc((size_t)BATCH * TDIM * FCD * 2);
    _Float16* zdyn_h   = (_Float16*)alloc((size_t)BATCH * TDIM * FCD * 2);
    // transposed f16 weights [N,K]
    _Float16* enc_wT   = (_Float16*)alloc((size_t)FCD * FEAT * 2);
    _Float16* gru_wihT = (_Float16*)alloc((size_t)3 * FCD * FCD * 2);
    _Float16* gru_whhT = (_Float16*)alloc((size_t)3 * FCD * FCD * 2);
    _Float16* sfe_wT   = (_Float16*)alloc((size_t)FCD * FCD * 2);
    _Float16* fmean_wT = (_Float16*)alloc((size_t)FDD * FCD * 2);
    _Float16* flogv_wT = (_Float16*)alloc((size_t)FDD * FCD * 2);
    _Float16* lstm_wihT= (_Float16*)alloc((size_t)4 * FCD * FCD * 2);
    _Float16* lstm_whhT= (_Float16*)alloc((size_t)4 * FCD * FCD * 2);
    _Float16* zmean_wT = (_Float16*)alloc((size_t)ZDD * FCD * 2);
    _Float16* zlogv_wT = (_Float16*)alloc((size_t)ZDD * FCD * 2);
    _Float16* p1_wihT  = (_Float16*)alloc((size_t)4 * HIDD * ZDD * 2);
    _Float16* p1_whhT  = (_Float16*)alloc((size_t)4 * HIDD * HIDD * 2);
    _Float16* p2_wihT  = (_Float16*)alloc((size_t)4 * HIDD * HIDD * 2);
    _Float16* p2_whhT  = (_Float16*)alloc((size_t)4 * HIDD * HIDD * 2);
    _Float16* pmean_wT = (_Float16*)alloc((size_t)ZDD * HIDD * 2);
    _Float16* plogv_wT = (_Float16*)alloc((size_t)ZDD * HIDD * 2);
    _Float16* decm_wT  = (_Float16*)alloc((size_t)FEAT * (ZDD + FDD) * 2);
    _Float16* decl_wT  = (_Float16*)alloc((size_t)FEAT * (ZDD + FDD) * 2);
    // activations / scratch
    float*    gi_big   = (float*)alloc((size_t)BATCH * NWW * 4 * FCD * 4);
    float*    gh_big   = (float*)alloc((size_t)BATCH * NWW * 4 * FCD * 4);
    float*    h_lstm   = (float*)alloc((size_t)BATCH * NWW * FCD * 4);
    float*    c_lstm   = (float*)alloc((size_t)BATCH * NWW * FCD * 4);
    _Float16* h_lstm_h = (_Float16*)alloc((size_t)BATCH * NWW * FCD * 2);
    float*    gi_gru   = (float*)alloc((size_t)BATCH * 3 * FCD * 4);
    float*    gh_gru   = (float*)alloc((size_t)BATCH * 3 * FCD * 4);
    float*    h_gru    = (float*)alloc((size_t)BATCH * FCD * 4);
    _Float16* h_gru_h  = (_Float16*)alloc((size_t)BATCH * FCD * 2);
    _Float16* f_h      = (_Float16*)alloc((size_t)BATCH * FCD * 2);
    _Float16* fpost_h  = (_Float16*)alloc((size_t)BATCH * FDD * 2);
    _Float16* ff_h     = (_Float16*)alloc((size_t)BATCH * FCD * 2);
    _Float16* zin_h    = (_Float16*)alloc((size_t)BATCH * NWW * ZDD * 2);
    float*    g1i      = (float*)alloc((size_t)BATCH * 4 * HIDD * 4);
    float*    g1h      = (float*)alloc((size_t)BATCH * 4 * HIDD * 4);
    float*    g2i      = (float*)alloc((size_t)BATCH * 4 * HIDD * 4);
    float*    g2h      = (float*)alloc((size_t)BATCH * 4 * HIDD * 4);
    float*    h1       = (float*)alloc((size_t)BATCH * HIDD * 4);
    float*    c1       = (float*)alloc((size_t)BATCH * HIDD * 4);
    float*    h2       = (float*)alloc((size_t)BATCH * HIDD * 4);
    float*    c2       = (float*)alloc((size_t)BATCH * HIDD * 4);
    _Float16* h1_h     = (_Float16*)alloc((size_t)BATCH * HIDD * 2);
    _Float16* h2_h     = (_Float16*)alloc((size_t)BATCH * HIDD * 2);
    _Float16* zf_h     = (_Float16*)alloc((size_t)BATCH * NWW * (ZDD + FDD) * 2);
    _Float16* zfs_h    = (_Float16*)alloc((size_t)BATCH * (ZDD + FDD) * 2);
    float*    dl_buf   = (float*)alloc((size_t)BATCH * NWW * FEAT * 4);
    float*    dls_buf  = (float*)alloc((size_t)BATCH * FEAT * 4);

    auto ew = [&](int n) { return dim3(cdiv(n, 256)); };
    auto gemm = [&](const _Float16* A, int lda, const _Float16* Bt, int ldb,
                    float* C, int ldc, _Float16* Ch, int ldch, const float* bias,
                    int M, int N, int K, int act) {
        dim3 g(cdiv(N, 64), M / 128);   // all GEMMs here have M % 128 == 0
        k_wmma_gemm<<<g, 256, 0, stream>>>(A, lda, Bt, ldb, C, ldc, Ch, ldch, bias,
                                           M, N, K, act);
    };
    auto xpose = [&](const float* W, _Float16* WT, int K, int N) {
        k_transpose_f16<<<ew(K * N), 256, 0, stream>>>(W, WT, K, N);
    };

    // ---- phase 0: conversions + weight transposes ----
    k_f32_to_f16<<<ew(BATCH * TDIM * FEAT), 256, 0, stream>>>(x_seq, xseq_h,
                                                              BATCH * TDIM * FEAT);
    xpose(enc_w, enc_wT, FEAT, FCD);
    xpose(gru_wih, gru_wihT, FCD, 3 * FCD);   xpose(gru_whh, gru_whhT, FCD, 3 * FCD);
    xpose(sfe_w, sfe_wT, FCD, FCD);
    xpose(fmean_w, fmean_wT, FCD, FDD);       xpose(flogvar_w, flogv_wT, FCD, FDD);
    xpose(lstm_wih, lstm_wihT, FCD, 4 * FCD); xpose(lstm_whh, lstm_whhT, FCD, 4 * FCD);
    xpose(zmean_w, zmean_wT, FCD, ZDD);       xpose(zlogvar_w, zlogv_wT, FCD, ZDD);
    xpose(p1_wih, p1_wihT, ZDD, 4 * HIDD);    xpose(p1_whh, p1_whhT, HIDD, 4 * HIDD);
    xpose(p2_wih, p2_wihT, HIDD, 4 * HIDD);   xpose(p2_whh, p2_whhT, HIDD, 4 * HIDD);
    xpose(pmean_w, pmean_wT, HIDD, ZDD);      xpose(plogvar_w, plogv_wT, HIDD, ZDD);
    xpose(decm_w, decm_wT, ZDD + FDD, FEAT);  xpose(decl_w, decl_wT, ZDD + FDD, FEAT);

    // ---- phase 1: encoder  fc_x = tanh(x @ enc_w + b)  (f16 mirror only) ----
    gemm(xseq_h, FEAT, enc_wT, FEAT, nullptr, 0, fcx_h, FCD, enc_b,
         BATCH * TDIM, FCD, FEAT, 1);

    // ---- phase 2: GRU over first window ----
    (void)hipMemsetAsync(h_gru, 0, (size_t)BATCH * FCD * 4, stream);
    (void)hipMemsetAsync(h_gru_h, 0, (size_t)BATCH * FCD * 2, stream);
    for (int t = 0; t < WINW; ++t) {
        gemm(fcx_h + (size_t)t * FCD, TDIM * FCD, gru_wihT, FCD,
             gi_gru, 3 * FCD, nullptr, 0, gru_bih, BATCH, 3 * FCD, FCD, 0);
        gemm(h_gru_h, FCD, gru_whhT, FCD,
             gh_gru, 3 * FCD, nullptr, 0, gru_bhh, BATCH, 3 * FCD, FCD, 0);
        k_gru_gate<<<ew(BATCH * FCD), 256, 0, stream>>>(gi_gru, gh_gru, h_gru,
                                                        h_gru, h_gru_h, BATCH, FCD);
    }

    // ---- phase 3: static feature  f = tanh(h @ sfe_w + b); f_mean/f_logvar/f_post ----
    gemm(h_gru_h, FCD, sfe_wT, FCD, nullptr, 0, f_h, FCD, sfe_b, BATCH, FCD, FCD, 1);
    gemm(f_h, FCD, fmean_wT, FCD, out + o_fmean, FDD, nullptr, 0, fmean_b,
         BATCH, FDD, FCD, 0);
    gemm(f_h, FCD, flogv_wT, FCD, out + o_flogvar, FDD, nullptr, 0, flogvar_b,
         BATCH, FDD, FCD, 0);
    k_sample<<<ew(BATCH * FDD), 256, 0, stream>>>(out + o_fmean, out + o_flogvar,
                                                  out + o_fpost, BATCH, FDD, FDD,
                                                  0x1111u);
    k_f32_to_f16<<<ew(BATCH * FDD), 256, 0, stream>>>(out + o_fpost, fpost_h,
                                                      BATCH * FDD);

    // ---- phase 4: dynamic branch per-window LSTM (merged batch B*NW, 8 steps) ----
    k_rng_normal_f16<<<ew(BATCH * FCD), 256, 0, stream>>>(ff_h, BATCH * FCD, 0x2222u);
    k_build_zdyn<<<ew(BATCH * TDIM * FCD), 256, 0, stream>>>(fcx_h, ff_h, zdyn_h,
                                                             BATCH * TDIM * FCD);
    (void)hipMemsetAsync(h_lstm, 0, (size_t)BATCH * NWW * FCD * 4, stream);
    (void)hipMemsetAsync(c_lstm, 0, (size_t)BATCH * NWW * FCD * 4, stream);
    (void)hipMemsetAsync(h_lstm_h, 0, (size_t)BATCH * NWW * FCD * 2, stream);
    for (int s = 0; s < WINW; ++s) {
        gemm(zdyn_h + (size_t)s * FCD, WINW * FCD, lstm_wihT, FCD,
             gi_big, 4 * FCD, nullptr, 0, lstm_bih, BATCH * NWW, 4 * FCD, FCD, 0);
        gemm(h_lstm_h, FCD, lstm_whhT, FCD,
             gh_big, 4 * FCD, nullptr, 0, lstm_bhh, BATCH * NWW, 4 * FCD, FCD, 0);
        k_lstm_gate<<<ew(BATCH * NWW * FCD), 256, 0, stream>>>(
            gi_big, gh_big, c_lstm, h_lstm, h_lstm_h, BATCH * NWW, FCD);
    }

    // ---- phase 5: z_mean / z_logvar / z_post ----
    gemm(h_lstm_h, FCD, zmean_wT, FCD, out + o_zmean, ZDD, nullptr, 0, zmean_b,
         BATCH * NWW, ZDD, FCD, 0);
    gemm(h_lstm_h, FCD, zlogv_wT, FCD, out + o_zlogvar, ZDD, nullptr, 0, zlogvar_b,
         BATCH * NWW, ZDD, FCD, 0);
    k_sample<<<ew(BATCH * NWW * ZDD), 256, 0, stream>>>(
        out + o_zmean, out + o_zlogvar, out + o_zpost, BATCH * NWW, ZDD, ZDD, 0x3333u);
    k_build_zin<<<ew(BATCH * NWW * ZDD), 256, 0, stream>>>(out + o_zpost, zin_h,
                                                           BATCH * NWW * ZDD);

    // ---- phase 6: motion prior (2-layer LSTMCell, 16 steps, teacher-forced) ----
    (void)hipMemsetAsync(h1, 0, (size_t)BATCH * HIDD * 4, stream);
    (void)hipMemsetAsync(c1, 0, (size_t)BATCH * HIDD * 4, stream);
    (void)hipMemsetAsync(h2, 0, (size_t)BATCH * HIDD * 4, stream);
    (void)hipMemsetAsync(c2, 0, (size_t)BATCH * HIDD * 4, stream);
    (void)hipMemsetAsync(h1_h, 0, (size_t)BATCH * HIDD * 2, stream);
    (void)hipMemsetAsync(h2_h, 0, (size_t)BATCH * HIDD * 2, stream);
    for (int t = 0; t < NWW; ++t) {
        gemm(zin_h + (size_t)t * ZDD, NWW * ZDD, p1_wihT, ZDD,
             g1i, 4 * HIDD, nullptr, 0, p1_bih, BATCH, 4 * HIDD, ZDD, 0);
        gemm(h1_h, HIDD, p1_whhT, HIDD,
             g1h, 4 * HIDD, nullptr, 0, p1_bhh, BATCH, 4 * HIDD, HIDD, 0);
        k_lstm_gate<<<ew(BATCH * HIDD), 256, 0, stream>>>(g1i, g1h, c1, h1, h1_h,
                                                          BATCH, HIDD);
        gemm(h1_h, HIDD, p2_wihT, HIDD,
             g2i, 4 * HIDD, nullptr, 0, p2_bih, BATCH, 4 * HIDD, HIDD, 0);
        gemm(h2_h, HIDD, p2_whhT, HIDD,
             g2h, 4 * HIDD, nullptr, 0, p2_bhh, BATCH, 4 * HIDD, HIDD, 0);
        k_lstm_gate<<<ew(BATCH * HIDD), 256, 0, stream>>>(g2i, g2h, c2, h2, h2_h,
                                                          BATCH, HIDD);
        gemm(h2_h, HIDD, pmean_wT, HIDD, out + o_zpm + t * ZDD, NWW * ZDD,
             nullptr, 0, pmean_b, BATCH, ZDD, HIDD, 0);
        gemm(h2_h, HIDD, plogv_wT, HIDD, out + o_zplv + t * ZDD, NWW * ZDD,
             nullptr, 0, plogvar_b, BATCH, ZDD, HIDD, 0);
        k_sample<<<ew(BATCH * ZDD), 256, 0, stream>>>(
            out + o_zpm + t * ZDD, out + o_zplv + t * ZDD, out + o_zp + t * ZDD,
            BATCH, ZDD, NWW * ZDD, 0x4000u + (uint32_t)t);
    }

    // ---- phase 7: Gaussian decoder ----
    k_build_zf<<<ew(BATCH * NWW * (ZDD + FDD)), 256, 0, stream>>>(
        out + o_zpost, fpost_h, zf_h, BATCH * NWW * (ZDD + FDD));
    gemm(zf_h, ZDD + FDD, decm_wT, ZDD + FDD, out + o_recon, FEAT, nullptr, 0,
         decm_b, BATCH * NWW, FEAT, ZDD + FDD, 0);
    gemm(zf_h, ZDD + FDD, decl_wT, ZDD + FDD, dl_buf, FEAT, nullptr, 0,
         decl_b, BATCH * NWW, FEAT, ZDD + FDD, 0);
    k_sample<<<ew(BATCH * NWW * FEAT), 256, 0, stream>>>(
        out + o_recon, dl_buf, out + o_recon, BATCH * NWW, FEAT, FEAT, 0x5555u);

    k_build_zf_stat<<<ew(BATCH * (ZDD + FDD)), 256, 0, stream>>>(
        out + o_zpost, fpost_h, zfs_h, BATCH * (ZDD + FDD));
    gemm(zfs_h, ZDD + FDD, decm_wT, ZDD + FDD, out + o_reconf, FEAT, nullptr, 0,
         decm_b, BATCH, FEAT, ZDD + FDD, 0);
    gemm(zfs_h, ZDD + FDD, decl_wT, ZDD + FDD, dls_buf, FEAT, nullptr, 0,
         decl_b, BATCH, FEAT, ZDD + FDD, 0);
    k_sample<<<ew(BATCH * FEAT), 256, 0, stream>>>(
        out + o_reconf, dls_buf, out + o_reconf, BATCH, FEAT, FEAT, 0x6666u);
}
